// BatchAllTtripletLoss_17016660426841
// MI455X (gfx1250) — matrix operand order repaired and checked by reading, not compiled, not run on hardware
//
#include <hip/hip_runtime.h>
#include <hip/hip_bf16.h>

typedef _Float16 v8h  __attribute__((ext_vector_type(8)));
typedef _Float16 v16h __attribute__((ext_vector_type(16)));
typedef float    v8f  __attribute__((ext_vector_type(8)));

#define BN     512
#define DDIM   768
#define FEPS   1e-8f
#define MARGIN 0.5f

// ---------------------------------------------------------------------------
// Kernel 1: per-row L2 normalize, fp32 -> f16 (WMMA input precision).
// One 256-thread block per row (8 waves on CDNA5 wave32).
// ---------------------------------------------------------------------------
__global__ void normalize_kernel(const float* __restrict__ x,
                                 _Float16* __restrict__ xn) {
    __shared__ float red[256];
    const int row = blockIdx.x;
    const int t   = threadIdx.x;
    const float* xr = x + row * DDIM;

    float ss = 0.0f;
    for (int c = t; c < DDIM; c += 256) { float v = xr[c]; ss += v * v; }
    red[t] = ss;
    __syncthreads();
    for (int s = 128; s > 0; s >>= 1) {
        if (t < s) red[t] += red[t + s];
        __syncthreads();
    }
    const float inv = 1.0f / fmaxf(sqrtf(red[0]), FEPS);  // torch cosine_sim semantics
    for (int c = t; c < DDIM; c += 256) {
        xn[row * DDIM + c] = (_Float16)(xr[c] * inv);
    }
}

// ---------------------------------------------------------------------------
// Kernel 2: cosine-distance matrix via v_wmma_f32_16x16x32_f16.
// One wave per 16x16 output tile; K-loop over 768 in steps of 32.
// dist[i][j] = 1 - xn_i . xn_j
// ---------------------------------------------------------------------------
__global__ void gram_wmma_kernel(const _Float16* __restrict__ xn,
                                 float* __restrict__ dist) {
    const int lane = threadIdx.x;        // 0..31, one wave per block
    const int g    = lane >> 4;          // lane group (0/1)
    const int mn   = lane & 15;          // row-within-tile (A) / col-within-tile (B)
    const int tm   = blockIdx.x;         // M tile
    const int tn   = blockIdx.y;         // N tile

    // A fragment source: row (tm*16+mn); B fragment source: row (tn*16+mn)
    // (B matrix is Xn^T, so column n of B == row n of Xn).
    const _Float16* arow = xn + (tm * 16 + mn) * DDIM;
    const _Float16* brow = xn + (tn * 16 + mn) * DDIM;

    v8f c = {};
    for (int k0 = 0; k0 < DDIM; k0 += 32) {
        // A 16x32 f16 layout: lane group g holds K = {8g..8g+7, 16+8g..16+8g+7}
        v8h alo = *(const v8h*)(arow + k0 + 8 * g);
        v8h ahi = *(const v8h*)(arow + k0 + 16 + 8 * g);
        // B 32x16 f16 layout: lane group g holds K = {8g..8g+7, 16+8g..16+8g+7}
        v8h blo = *(const v8h*)(brow + k0 + 8 * g);
        v8h bhi = *(const v8h*)(brow + k0 + 16 + 8 * g);

        v16h a, b;
#pragma unroll
        for (int t = 0; t < 8; ++t) {
            a[t] = alo[t]; a[t + 8] = ahi[t];
            b[t] = blo[t]; b[t + 8] = bhi[t];
        }
        // 8 args: (neg_a, A, neg_b, B, c_mod, C, reuse_a, reuse_b)
        c = __builtin_amdgcn_wmma_f32_16x16x32_f16(
                false, a, false, b, (short)0, c, false, false);
    }

    // C/D layout: VGPR r, lane group g -> (row r+8g, col mn) within tile.
    const int col = tn * 16 + mn;
#pragma unroll
    for (int r = 0; r < 8; ++r) {
        const int row = tm * 16 + r + 8 * g;
        dist[row * BN + col] = 1.0f - c[r];
    }
}

// ---------------------------------------------------------------------------
// Kernel 3a: zero the accumulators (float sum, uint positive-count).
// ---------------------------------------------------------------------------
__global__ void init_accum_kernel(float* accum) {
    accum[0] = 0.0f;
    ((unsigned*)accum)[1] = 0u;
}

// ---------------------------------------------------------------------------
// Kernel 3b: batch-all triplet reduction.
// Thread <-> (i,j) pair; i is constant per 256-thread block (256 | 512), so
// dist row i and all labels are staged in LDS. Valid triplet mask collapses
// to: i!=j && lab[i]==lab[j] && lab[i]!=lab[k] (label inequality implies
// k!=i and k!=j).
// ---------------------------------------------------------------------------
__global__ void triplet_reduce_kernel(const float* __restrict__ dist,
                                      const int* __restrict__ labels,
                                      float* __restrict__ accum) {
    __shared__ float    drow[BN];
    __shared__ int      lab[BN];
    __shared__ float    rsum[256];
    __shared__ unsigned rcnt[256];

    const int t = threadIdx.x;
    const int p = blockIdx.x * 256 + t;
    const int i = p >> 9;            // same for all threads in block
    const int j = p & (BN - 1);

    for (int c = t; c < BN; c += 256) {
        drow[c] = dist[i * BN + c];
        lab[c]  = labels[c];
    }
    __syncthreads();

    float    sum = 0.0f;
    unsigned cnt = 0u;
    const int li = lab[i];
    if (i != j && li == lab[j]) {
        const float dij = drow[j];
        for (int k = 0; k < BN; ++k) {
            if (lab[k] != li) {
                float v = dij - drow[k] + MARGIN;
                v = fmaxf(v, 0.0f);
                sum += v;
                cnt += (v > FEPS) ? 1u : 0u;
            }
        }
    }

    rsum[t] = sum;
    rcnt[t] = cnt;
    __syncthreads();
    for (int s = 128; s > 0; s >>= 1) {
        if (t < s) { rsum[t] += rsum[t + s]; rcnt[t] += rcnt[t + s]; }
        __syncthreads();
    }
    if (t == 0) {
        atomicAdd(accum, rsum[0]);
        atomicAdd(((unsigned*)accum) + 1, rcnt[0]);
    }
}

// ---------------------------------------------------------------------------
// Kernel 4: final scalar  sum / (num_pos + eps)
// ---------------------------------------------------------------------------
__global__ void finalize_kernel(const float* __restrict__ accum,
                                float* __restrict__ out) {
    out[0] = accum[0] / ((float)((const unsigned*)accum)[1] + FEPS);
}

// ---------------------------------------------------------------------------
// Workspace layout (all 16B-aligned):
//   [0,        786432)  Xn   f16  512x768
//   [786432, 1835008)   dist f32  512x512
//   [1835008, 1835016)  accum {float sum, uint count}
// ---------------------------------------------------------------------------
extern "C" void kernel_launch(void* const* d_in, const int* in_sizes, int n_in,
                              void* d_out, int out_size, void* d_ws, size_t ws_size,
                              hipStream_t stream) {
    (void)in_sizes; (void)n_in; (void)out_size; (void)ws_size;

    const float* emb    = (const float*)d_in[0];
    const int*   labels = (const int*)d_in[1];   // jax default int32 (x64 disabled)
    float*       out    = (float*)d_out;

    char*      ws    = (char*)d_ws;
    _Float16*  xn    = (_Float16*)ws;
    float*     dist  = (float*)(ws + 786432);
    float*     accum = (float*)(ws + 786432 + 1048576);

    normalize_kernel<<<BN, 256, 0, stream>>>(emb, xn);
    gram_wmma_kernel<<<dim3(BN / 16, BN / 16), 32, 0, stream>>>(xn, dist);
    init_accum_kernel<<<1, 1, 0, stream>>>(accum);
    triplet_reduce_kernel<<<(BN * BN) / 256, 256, 0, stream>>>(dist, labels, accum);
    finalize_kernel<<<1, 1, 0, stream>>>(accum, out);
}
